// GenomicLSTM_65309272703613
// MI455X (gfx1250) — matrix operand order repaired and checked by reading, not compiled, not run on hardware
//
#include <hip/hip_runtime.h>
#include <math.h>

// Fused 2-layer LSTM + FC for (B=512, T=4096, I=8, H=64) on gfx1250 (CDNA5).
//
// One workgroup per 16-batch tile; gate weights live in VGPRs as WMMA
// B-fragments for the whole 4096-step scan; h/c live in LDS; both layers
// fused per timestep so h1 never touches HBM (only x is streamed, prefetched
// one step ahead). Matrix work: v_wmma_f32_16x16x32_f16 (14 per wave per
// step). Activations: native CDNA5 v_tanh_f32. Elementwise phases fully
// unrolled (4 cells/thread) so independent tanh chains interleave.

typedef __attribute__((ext_vector_type(16))) _Float16 v16h;
typedef __attribute__((ext_vector_type(8)))  float    v8f;
typedef __attribute__((ext_vector_type(4)))  float    f32x4;

constexpr int BT  = 16;    // batch rows per workgroup
constexpr int HD  = 64;    // hidden
constexpr int TT  = 4096;  // timesteps
constexpr int IN  = 8;     // input features
constexpr int AS0 = 104;   // LDS row stride (halves) for A0 = [x(8)|pad(24)|h0(64)], K=96 padded
constexpr int AS1 = 136;   // LDS row stride (halves) for A1 = [h0(64)|h1(64)], K=128 padded
constexpr int GS  = 257;   // LDS row stride (floats) for gate scratch (odd -> no bank alias)
constexpr int NCELL = (BT * HD) / 256;  // elementwise cells per thread (= 4)

// ---- fast activations: single native transcendental each ----
#if __has_builtin(__builtin_amdgcn_tanhf)
__device__ __forceinline__ float fast_tanh(float x) { return __builtin_amdgcn_tanhf(x); }
#else
__device__ __forceinline__ float fast_tanh(float x) {
#if __has_builtin(__builtin_amdgcn_rcpf)
    return 1.0f - 2.0f * __builtin_amdgcn_rcpf(__expf(2.0f * x) + 1.0f);
#else
    return 1.0f - 2.0f / (__expf(2.0f * x) + 1.0f);
#endif
}
#endif
__device__ __forceinline__ float sigm(float x) { return fmaf(0.5f, fast_tanh(0.5f * x), 0.5f); }

// Combined layer-0 weight lookup over K=96: [W_ih0(8) | zeros(24) | W_hh0(64)]
__device__ __forceinline__ float wcat0(const float* Wih, const float* Whh, int n, int k) {
    if (k < 8)  return Wih[n * 8 + k];
    if (k < 32) return 0.0f;
    return Whh[n * 64 + (k - 32)];
}
// Combined layer-1 weight lookup over K=128: [W_ih1(64) | W_hh1(64)]
__device__ __forceinline__ float wcat1(const float* Wih, const float* Whh, int n, int k) {
    return (k < 64) ? Wih[n * 64 + k] : Whh[n * 64 + (k - 64)];
}

// Load a 16x32 f16 A-fragment from LDS (ISA 7.12.2 layout):
// lane L (row = L%16): K = k0 + (L<16 ? 0..7 : 8..15) in v[0..3],
//                      K = k0 + (L<16 ? 16..23 : 24..31) in v[4..7].
// Two 16B ds_load_b128 per lane; strides chosen conflict-free.
__device__ __forceinline__ v16h ldA(const _Float16* abuf, int strideH, int k0, int lane) {
    int row = lane & 15;
    int off = (lane & 16) ? 8 : 0;
    const _Float16* p = abuf + row * strideH + k0 + off;
    union { v16h v; f32x4 q[2]; } u;
    u.q[0] = *(const f32x4*)(p);
    u.q[1] = *(const f32x4*)(p + 16);
    return u.v;
}

__global__ __launch_bounds__(256)
void lstm2_fused_wmma(const float* __restrict__ x,
                      const float* __restrict__ Wih0, const float* __restrict__ Whh0,
                      const float* __restrict__ bih0, const float* __restrict__ bhh0,
                      const float* __restrict__ Wih1, const float* __restrict__ Whh1,
                      const float* __restrict__ bih1, const float* __restrict__ bhh1,
                      const float* __restrict__ fcw,  const float* __restrict__ fcb,
                      float* __restrict__ out) {
    __shared__ _Float16 sA0[BT * AS0];   // layer0 A operand  [x | 0 | h0]
    __shared__ _Float16 sA1[BT * AS1];   // layer1 A operand  [h0 | h1]
    __shared__ float    sG [BT * GS];    // gate scratch (one layer at a time)
    __shared__ float    sC0[BT * HD];
    __shared__ float    sC1[BT * HD];
    __shared__ float    sH1[BT * HD];    // f32 copy of h1 at t=T-1 for the FC

    const int tid   = threadIdx.x;
    const int lane  = tid & 31;
    const int wave  = tid >> 5;          // 8 waves, wave w owns gate cols [32w, 32w+32)
    const int bbase = blockIdx.x * BT;

    // ---- zero-init LDS state (h0 = h1 = c0 = c1 = 0) ----
    for (int i = tid; i < BT * AS0; i += 256) sA0[i] = (_Float16)0.0f;
    for (int i = tid; i < BT * AS1; i += 256) sA1[i] = (_Float16)0.0f;
    for (int i = tid; i < BT * HD;  i += 256) { sC0[i] = 0.0f; sC1[i] = 0.0f; }

    // ---- build per-wave resident weight B-fragments (one-time) ----
    // B-fragment layout (16x32 f16 B, KxN): lane L holds col N = ntile+L%16,
    // K = k0 + (L<16 ? 0..7,16..23 : 8..15,24..31), packed 2/VGPR (matches A).
    const int nloc = lane & 15;
    const int nt0  = wave * 32;
    const int nt1  = wave * 32 + 16;
    const int koff = (lane & 16) ? 16 : 0;

    v16h wB0[2][3];
    v16h wB1[2][4];
    for (int tle = 0; tle < 2; ++tle) {
        int n = (tle ? nt1 : nt0) + nloc;
        for (int c = 0; c < 3; ++c) {
            int kb = 32 * c + (koff >> 1);
            v16h f;
            for (int j = 0; j < 16; ++j) f[j] = (_Float16)wcat0(Wih0, Whh0, n, kb + j + (j < 8 ? 0 : 8));
            wB0[tle][c] = f;
        }
        for (int c = 0; c < 4; ++c) {
            int kb = 32 * c + (koff >> 1);
            v16h f;
            for (int j = 0; j < 16; ++j) f[j] = (_Float16)wcat1(Wih1, Whh1, n, kb + j + (j < 8 ? 0 : 8));
            wB1[tle][c] = f;
        }
    }
    const float bias0v[2] = { bih0[nt0 + nloc] + bhh0[nt0 + nloc],
                              bih0[nt1 + nloc] + bhh0[nt1 + nloc] };
    const float bias1v[2] = { bih1[nt0 + nloc] + bhh1[nt0 + nloc],
                              bih1[nt1 + nloc] + bhh1[nt1 + nloc] };

    // ---- x streaming: thread tid<128 owns (b = tid/8, d = tid%8) ----
    float  xreg = 0.0f;
    size_t xbase = 0;
    const bool xowner = (tid < BT * IN);
    if (xowner) {
        int b = tid >> 3, d = tid & 7;
        xbase = ((size_t)(bbase + b) * TT) * IN + d;
        xreg  = x[xbase];                            // t = 0
        sA0[b * AS0 + d] = (_Float16)xreg;           // stage x_0
    }
    __syncthreads();
    if (xowner) xreg = x[xbase + IN];                // prefetch x_1

    const int rb = (lane & 16) ? 8 : 0;              // accumulator row base
    const int eb = tid >> 6;                         // elementwise base batch row
    const int ej = tid & 63;                         // elementwise hidden col

#pragma unroll 1
    for (int t = 0; t < TT; ++t) {
        // ---- layer 0: gates = [x|0|h0] (16x96) @ Wcat0^T -> 16x256 ----
        {
            v16h a0 = ldA(sA0, AS0, 0,  lane);
            v16h a1 = ldA(sA0, AS0, 32, lane);
            v16h a2 = ldA(sA0, AS0, 64, lane);
            v8f acc0, acc1;
            for (int r = 0; r < 8; ++r) { acc0[r] = bias0v[0]; acc1[r] = bias0v[1]; }
            acc0 = __builtin_amdgcn_wmma_f32_16x16x32_f16(false, a0, false, wB0[0][0], (short)0, acc0, false, false);
            acc1 = __builtin_amdgcn_wmma_f32_16x16x32_f16(false, a0, false, wB0[1][0], (short)0, acc1, false, false);
            acc0 = __builtin_amdgcn_wmma_f32_16x16x32_f16(false, a1, false, wB0[0][1], (short)0, acc0, false, false);
            acc1 = __builtin_amdgcn_wmma_f32_16x16x32_f16(false, a1, false, wB0[1][1], (short)0, acc1, false, false);
            acc0 = __builtin_amdgcn_wmma_f32_16x16x32_f16(false, a2, false, wB0[0][2], (short)0, acc0, false, false);
            acc1 = __builtin_amdgcn_wmma_f32_16x16x32_f16(false, a2, false, wB0[1][2], (short)0, acc1, false, false);
            for (int r = 0; r < 8; ++r) {
                sG[(rb + r) * GS + nt0 + nloc] = acc0[r];
                sG[(rb + r) * GS + nt1 + nloc] = acc1[r];
            }
        }
        __syncthreads();

        // ---- layer 0 elementwise (fully unrolled, 4 independent cells) ----
#pragma unroll
        for (int it = 0; it < NCELL; ++it) {
            int b   = eb + it * 4;           // rows eb, eb+4, eb+8, eb+12
            int idx = b * HD + ej;
            const float* g = &sG[b * GS + ej];
            float i_ = sigm     (g[0]);
            float f_ = sigm     (g[64]);
            float g_ = fast_tanh(g[128]);
            float o_ = sigm     (g[192]);
            float c  = fmaf(f_, sC0[idx], i_ * g_);
            sC0[idx] = c;
            float h  = o_ * fast_tanh(c);
            sA0[b * AS0 + 32 + ej] = (_Float16)h;   // h0 for next step's layer0
            sA1[b * AS1 + ej]      = (_Float16)h;   // h0 for this step's layer1
        }
        __syncthreads();

        // ---- layer 1: gates = [h0_t|h1_{t-1}] (16x128) @ Wcat1^T ----
        {
            v16h a0 = ldA(sA1, AS1, 0,  lane);
            v16h a1 = ldA(sA1, AS1, 32, lane);
            v16h a2 = ldA(sA1, AS1, 64, lane);
            v16h a3 = ldA(sA1, AS1, 96, lane);
            v8f acc0, acc1;
            for (int r = 0; r < 8; ++r) { acc0[r] = bias1v[0]; acc1[r] = bias1v[1]; }
            acc0 = __builtin_amdgcn_wmma_f32_16x16x32_f16(false, a0, false, wB1[0][0], (short)0, acc0, false, false);
            acc1 = __builtin_amdgcn_wmma_f32_16x16x32_f16(false, a0, false, wB1[1][0], (short)0, acc1, false, false);
            acc0 = __builtin_amdgcn_wmma_f32_16x16x32_f16(false, a1, false, wB1[0][1], (short)0, acc0, false, false);
            acc1 = __builtin_amdgcn_wmma_f32_16x16x32_f16(false, a1, false, wB1[1][1], (short)0, acc1, false, false);
            acc0 = __builtin_amdgcn_wmma_f32_16x16x32_f16(false, a2, false, wB1[0][2], (short)0, acc0, false, false);
            acc1 = __builtin_amdgcn_wmma_f32_16x16x32_f16(false, a2, false, wB1[1][2], (short)0, acc1, false, false);
            acc0 = __builtin_amdgcn_wmma_f32_16x16x32_f16(false, a3, false, wB1[0][3], (short)0, acc0, false, false);
            acc1 = __builtin_amdgcn_wmma_f32_16x16x32_f16(false, a3, false, wB1[1][3], (short)0, acc1, false, false);
            for (int r = 0; r < 8; ++r) {
                sG[(rb + r) * GS + nt0 + nloc] = acc0[r];
                sG[(rb + r) * GS + nt1 + nloc] = acc1[r];
            }
        }
        __syncthreads();

        // ---- layer 1 elementwise + stage x_{t+1} (same barrier section) ----
#pragma unroll
        for (int it = 0; it < NCELL; ++it) {
            int b   = eb + it * 4;
            int idx = b * HD + ej;
            const float* g = &sG[b * GS + ej];
            float i_ = sigm     (g[0]);
            float f_ = sigm     (g[64]);
            float g_ = fast_tanh(g[128]);
            float o_ = sigm     (g[192]);
            float c  = fmaf(f_, sC1[idx], i_ * g_);
            sC1[idx] = c;
            float h  = o_ * fast_tanh(c);
            sA1[b * AS1 + 64 + ej] = (_Float16)h;   // h1 for next step
            if (t == TT - 1) sH1[idx] = h;           // uniform branch, last step only
        }
        if (xowner && t + 1 < TT)
            sA0[(tid >> 3) * AS0 + (tid & 7)] = (_Float16)xreg;   // stage x_{t+1}
        __syncthreads();
        if (xowner && t + 2 < TT)
            xreg = x[xbase + (size_t)(t + 2) * IN];               // prefetch x_{t+2}
    }

    // ---- final FC + sigmoid on h1 at t = T-1 ----
    if (tid < BT) {
        float s = fcb[0];
        for (int j = 0; j < HD; ++j) s += sH1[tid * HD + j] * fcw[j];
        out[bbase + tid] = sigm(s);
    }
}

extern "C" void kernel_launch(void* const* d_in, const int* in_sizes, int n_in,
                              void* d_out, int out_size, void* d_ws, size_t ws_size,
                              hipStream_t stream) {
    (void)in_sizes; (void)n_in; (void)out_size; (void)d_ws; (void)ws_size;
    const float* x    = (const float*)d_in[0];
    const float* Wih0 = (const float*)d_in[1];
    const float* Whh0 = (const float*)d_in[2];
    const float* bih0 = (const float*)d_in[3];
    const float* bhh0 = (const float*)d_in[4];
    const float* Wih1 = (const float*)d_in[5];
    const float* Whh1 = (const float*)d_in[6];
    const float* bih1 = (const float*)d_in[7];
    const float* bhh1 = (const float*)d_in[8];
    const float* fcw  = (const float*)d_in[9];
    const float* fcb  = (const float*)d_in[10];
    float* out = (float*)d_out;

    dim3 grid(512 / BT);   // 32 workgroups
    dim3 block(256);       // 8 waves, wave w owns gate columns [32w, 32w+32)
    lstm2_fused_wmma<<<grid, block, 0, stream>>>(x, Wih0, Whh0, bih0, bhh0,
                                                 Wih1, Whh1, bih1, bhh1,
                                                 fcw, fcb, out);
}
